// EnhancedGNNPredictor_81784767250570
// MI455X (gfx1250) — matrix operand order repaired and checked by reading, not compiled, not run on hardware
//
#include <hip/hip_runtime.h>
#include <math.h>

// ---------------------------------------------------------------------------
// EnhancedGNNPredictor (3-layer GATv2, N=100000, E=800000, D=128, H=4) on
// MI455X / gfx1250.  GEMMs via v_wmma_f32_16x16x32_bf16 (2 M-tiles per wave,
// B-fragment reuse); edge phase exploits the 192MB L2 (xl/xr tables are
// 51.2MB each -> L2 resident).
// ---------------------------------------------------------------------------

typedef __attribute__((ext_vector_type(16))) __bf16 v16bf;
typedef __attribute__((ext_vector_type(8)))  __bf16 v8bf;
typedef __attribute__((ext_vector_type(8)))  float  v8f;

#define N_NODES 100000
#define N_EDGES 800000
#define DIM     128
#define HEADS   4
#define LAYERS  3

__device__ __forceinline__ float atomAddF(float* p, float v) {
  return __hip_atomic_fetch_add(p, v, __ATOMIC_RELAXED, __HIP_MEMORY_SCOPE_AGENT);
}
__device__ __forceinline__ void atomMaxF(float* p, float v) {
  (void)__hip_atomic_fetch_max(p, v, __ATOMIC_RELAXED, __HIP_MEMORY_SCOPE_AGENT);
}
__device__ __forceinline__ float waveRedSum(float v) {
#pragma unroll
  for (int off = 16; off; off >>= 1) v += __shfl_xor(v, off, 32);
  return v;
}

// --------------------------- f32 -> bf16 convert ---------------------------
__global__ void f32_to_bf16_kernel(const float* __restrict__ in,
                                   __bf16* __restrict__ out, int n) {
  int i = blockIdx.x * blockDim.x + threadIdx.x;
  if (i < n) out[i] = (__bf16)in[i];
}

// ------------------- weight prep: transpose + bf16 -------------------------
// Wt[n*128 + k] = W[k*128 + n]  (so a B-fragment column is contiguous in k)
__global__ void prep_weight_kernel(const float* __restrict__ W,
                                   __bf16* __restrict__ Wt) {
  int i = blockIdx.x * blockDim.x + threadIdx.x;   // 0 .. 16383
  int n = i & 127;
  int k = i >> 7;
  Wt[n * DIM + k] = (__bf16)W[k * DIM + n];
}

// ------------------------------ WMMA GEMM ----------------------------------
// out[M,128] = A_bf16[M,128] @ W[128,128] (+ bias), W given pre-transposed
// bf16.  Block = 256 threads = 8 waves; block b -> rows [32b,32b+32)
// (two 16-row M-tiles per wave, B fragments reused), wave w -> columns
// [16w,16w+16).  K-loop: 4 x (K=32), 8 WMMA per wave.  N%32==0 -> no guards,
// EXEC stays all-1s as WMMA requires.
__global__ __launch_bounds__(256)
void gemm_bf16_wmma_kernel(const __bf16* __restrict__ A,
                           const __bf16* __restrict__ Wt,
                           const float* __restrict__ bias,
                           float* __restrict__ out) {
  const int wave = threadIdx.x >> 5;       // 0..7 -> column tile
  const int lane = threadIdx.x & 31;
  const int half = lane >> 4;              // 0 | 1
  const int l16  = lane & 15;
  const int rowA0 = (blockIdx.x << 5) + l16;        // M-tile 0
  const int rowA1 = rowA0 + 16;                     // M-tile 1
  const int colB  = (wave << 4) + l16;

  v8f c0 = {0.f, 0.f, 0.f, 0.f, 0.f, 0.f, 0.f, 0.f};
  v8f c1 = {0.f, 0.f, 0.f, 0.f, 0.f, 0.f, 0.f, 0.f};

#pragma unroll
  for (int k0 = 0; k0 < DIM; k0 += 32) {
    // B fragment: lane<16 -> K k0+0..7 & k0+16..23 ; lane>=16 -> +8 (one column)
    const int kbB = half ? 8 : 0;
    v8bf b0 = *(const v8bf*)(Wt + (size_t)colB * DIM + k0 + kbB);
    v8bf b1 = *(const v8bf*)(Wt + (size_t)colB * DIM + k0 + kbB + 16);
    v16bf b;
#pragma unroll
    for (int e = 0; e < 8; ++e) { b[e] = b0[e]; b[8 + e] = b1[e]; }

    // A fragments: lane<16 -> K k0+0..7 & k0+16..23 ; lane>=16 -> +8
    const int kbA = half ? 8 : 0;
    v8bf a0lo = *(const v8bf*)(A + (size_t)rowA0 * DIM + k0 + kbA);
    v8bf a0hi = *(const v8bf*)(A + (size_t)rowA0 * DIM + k0 + 16 + kbA);
    v8bf a1lo = *(const v8bf*)(A + (size_t)rowA1 * DIM + k0 + kbA);
    v8bf a1hi = *(const v8bf*)(A + (size_t)rowA1 * DIM + k0 + 16 + kbA);
    v16bf a0, a1;
#pragma unroll
    for (int e = 0; e < 8; ++e) {
      a0[e] = a0lo[e]; a0[8 + e] = a0hi[e];
      a1[e] = a1lo[e]; a1[8 + e] = a1hi[e];
    }

    c0 = __builtin_amdgcn_wmma_f32_16x16x32_bf16(false, a0, false, b,
                                                 (short)0, c0, false, false);
    c1 = __builtin_amdgcn_wmma_f32_16x16x32_bf16(false, a1, false, b,
                                                 (short)0, c1, false, false);
  }

  const float bb = bias ? bias[colB] : 0.0f;
  const int rowBase0 = (blockIdx.x << 5) + (half ? 8 : 0);
#pragma unroll
  for (int r = 0; r < 8; ++r) {
    out[(size_t)(rowBase0 + r) * DIM + colB]      = c0[r] + bb;
    out[(size_t)(rowBase0 + 16 + r) * DIM + colB] = c1[r] + bb;
  }
}

// ----------------------------- layer init ----------------------------------
__global__ void init_layer_kernel(float* __restrict__ agg,
                                  float* __restrict__ lmax,
                                  float* __restrict__ denom,
                                  float* __restrict__ colsum, int nd, int nh) {
  int i = blockIdx.x * blockDim.x + threadIdx.x;
  if (i < nd) agg[i] = 0.0f;
  if (i < nh) { lmax[i] = -3.0e38f; denom[i] = 0.0f; }
  if (i < DIM) colsum[i] = 0.0f;
}

// ------------------- edge pass 1: logits + segment max ---------------------
// one wave per edge; lane covers channel `lane` of each of the 4 heads
__global__ __launch_bounds__(256)
void edge_logits_kernel(const int* __restrict__ src, const int* __restrict__ dst,
                        const float* __restrict__ xl, const float* __restrict__ xr,
                        const float* __restrict__ att,
                        float* __restrict__ logits, float* __restrict__ lmax) {
  const int wid = blockIdx.x * 8 + (threadIdx.x >> 5);
  if (wid >= N_EDGES) return;
  const int lane = threadIdx.x & 31;
  const int s = src[wid], d = dst[wid];

  float t[HEADS];
#pragma unroll
  for (int h = 0; h < HEADS; ++h) {
    const int f = h * 32 + lane;
    float v = xl[(size_t)s * DIM + f] + xr[(size_t)d * DIM + f];
    v = (v > 0.0f) ? v : 0.2f * v;           // leaky_relu(0.2)
    t[h] = v * att[f];
  }
#pragma unroll
  for (int h = 0; h < HEADS; ++h) t[h] = waveRedSum(t[h]);

  if (lane == 0) {
#pragma unroll
    for (int h = 0; h < HEADS; ++h) {
      logits[(size_t)wid * HEADS + h] = t[h];
      atomMaxF(&lmax[(size_t)d * HEADS + h], t[h]);
    }
  }
}

// --------- edge pass 2: a = exp(logit - max); denom & weighted agg ---------
__global__ __launch_bounds__(256)
void edge_aggregate_kernel(const int* __restrict__ src, const int* __restrict__ dst,
                           const float* __restrict__ xl,
                           const float* __restrict__ logits,
                           const float* __restrict__ lmax,
                           float* __restrict__ denom, float* __restrict__ agg) {
  const int wid = blockIdx.x * 8 + (threadIdx.x >> 5);
  if (wid >= N_EDGES) return;
  const int lane = threadIdx.x & 31;
  const int s = src[wid], d = dst[wid];

  const int hl = lane & 3;
  const float al = __expf(logits[(size_t)wid * HEADS + hl] -
                          lmax[(size_t)d * HEADS + hl]);
  if (lane < HEADS) atomAddF(&denom[(size_t)d * HEADS + lane], al);

#pragma unroll
  for (int h = 0; h < HEADS; ++h) {
    const float ah = __shfl(al, h, 32);
    const int f = h * 32 + lane;
    atomAddF(&agg[(size_t)d * DIM + f], ah * xl[(size_t)s * DIM + f]);
  }
}

// ---- node pass: divide by denom, +b_out, accumulate pairnorm column sum ----
__global__ __launch_bounds__(256)
void node_finalize_kernel(float* __restrict__ agg, const float* __restrict__ denom,
                          const float* __restrict__ b_out,
                          float* __restrict__ colsum) {
  __shared__ float ls[DIM];
  const int tid = threadIdx.x;
  if (tid < DIM) ls[tid] = 0.0f;
  __syncthreads();

  const int wid = blockIdx.x * 8 + (tid >> 5);
  const int lane = tid & 31;
  if (wid < N_NODES) {
#pragma unroll
    for (int h = 0; h < HEADS; ++h) {
      const int f = h * 32 + lane;
      float val = agg[(size_t)wid * DIM + f] /
                      (denom[(size_t)wid * HEADS + h] + 1e-16f) +
                  b_out[f];
      agg[(size_t)wid * DIM + f] = val;
      atomicAdd(&ls[f], val);                 // ds_add_f32
    }
  }
  __syncthreads();
  if (tid < DIM) atomAddF(&colsum[tid], ls[tid]);
}

// ---- fused: pairnorm + residual + layernorm + optional relu (wave/row) ----
__global__ __launch_bounds__(256)
void row_final_kernel(const float* __restrict__ agg, const float* __restrict__ colsum,
                      const float* ident, const float* __restrict__ ln_g,
                      const float* __restrict__ ln_b, float* outp, int do_relu) {
  const int wid = blockIdx.x * 8 + (threadIdx.x >> 5);
  if (wid >= N_NODES) return;
  const int lane = threadIdx.x & 31;
  const float invN = 1.0f / (float)N_NODES;

  float v[HEADS], idv[HEADS];
  float ss = 0.0f;
#pragma unroll
  for (int h = 0; h < HEADS; ++h) {
    const int f = h * 32 + lane;
    idv[h] = ident[(size_t)wid * DIM + f];       // read residual BEFORE write
    v[h] = agg[(size_t)wid * DIM + f] - colsum[f] * invN;
    ss += v[h] * v[h];
  }
  ss = waveRedSum(ss);
  const float pnScale = sqrtf((float)N_NODES) / (sqrtf(ss) + 1e-6f);

  float w[HEADS], mu = 0.0f;
#pragma unroll
  for (int h = 0; h < HEADS; ++h) { w[h] = v[h] * pnScale + idv[h]; mu += w[h]; }
  mu = waveRedSum(mu) * (1.0f / DIM);

  float var = 0.0f;
#pragma unroll
  for (int h = 0; h < HEADS; ++h) { float dvi = w[h] - mu; var += dvi * dvi; }
  var = waveRedSum(var) * (1.0f / DIM);
  const float inv = rsqrtf(var + 1e-5f);

#pragma unroll
  for (int h = 0; h < HEADS; ++h) {
    const int f = h * 32 + lane;
    float y = (w[h] - mu) * inv * ln_g[f] + ln_b[f];
    if (do_relu) y = fmaxf(y, 0.0f);
    outp[(size_t)wid * DIM + f] = y;
  }
}

// ---------------------------------------------------------------------------
extern "C" void kernel_launch(void* const* d_in, const int* in_sizes, int n_in,
                              void* d_out, int out_size, void* d_ws, size_t ws_size,
                              hipStream_t stream) {
  (void)in_sizes; (void)n_in; (void)out_size; (void)ws_size;

  const float* x      = (const float*)d_in[0];
  const int*   ei     = (const int*)d_in[1];
  const float* w_in   = (const float*)d_in[2];
  const float* b_in   = (const float*)d_in[3];
  const float* Wl     = (const float*)d_in[4];
  const float* bl     = (const float*)d_in[5];
  const float* Wr     = (const float*)d_in[6];
  const float* att    = (const float*)d_in[7];
  const float* b_out  = (const float*)d_in[8];
  const float* Wres   = (const float*)d_in[9];
  const float* bres   = (const float*)d_in[10];
  const float* ln_g   = (const float*)d_in[11];
  const float* ln_b   = (const float*)d_in[12];

  const int* srcp = ei;
  const int* dstp = ei + N_EDGES;

  const size_t ND  = (size_t)N_NODES * DIM;
  const size_t NH  = (size_t)N_NODES * HEADS;
  const size_t EH  = (size_t)N_EDGES * HEADS;
  const size_t DD2 = (size_t)DIM * DIM;

  // ------------------------ workspace carve-up -----------------------------
  float* ws       = (float*)d_ws;
  float* xlbuf    = ws;
  float* xrbuf    = xlbuf + ND;
  float* aggbuf   = xrbuf + ND;
  float* identbuf = aggbuf + ND;
  float* logitsb  = identbuf + ND;
  float* lmaxb    = logitsb + EH;
  float* denomb   = lmaxb + NH;
  float* colsumb  = denomb + NH;
  __bf16* abf     = (__bf16*)(colsumb + 256);        // [N,128] bf16 activations
  __bf16* wtbuf   = abf + ND;                        // 9 transposed bf16 weights
  __bf16* wt_in   = wtbuf;
  __bf16* wtL     = wt_in + DD2;                     // 3 matrices
  __bf16* wtR     = wtL + 3 * DD2;                   // 3 matrices
  __bf16* wtRes   = wtR + 3 * DD2;                   // 2 matrices

  float* hbuf = (float*)d_out;                       // h lives in d_out

  const int cvtBlocks  = (int)((ND + 255) / 256);
  const int gemmBlocks = N_NODES / 32;               // 3125 (two M-tiles/wave)
  const int edgeBlocks = N_EDGES / 8;                // 100000
  const int nodeBlocks = (N_NODES + 7) / 8;          // 12500

  // ------------------- weight prep (transpose + bf16) ----------------------
  prep_weight_kernel<<<64, 256, 0, stream>>>(w_in, wt_in);
  for (int l = 0; l < LAYERS; ++l) {
    prep_weight_kernel<<<64, 256, 0, stream>>>(Wl + l * DD2, wtL + l * DD2);
    prep_weight_kernel<<<64, 256, 0, stream>>>(Wr + l * DD2, wtR + l * DD2);
  }
  for (int l = 0; l < LAYERS - 1; ++l)
    prep_weight_kernel<<<64, 256, 0, stream>>>(Wres + l * DD2, wtRes + l * DD2);

  // --------------------------- h = x @ w_in + b_in -------------------------
  f32_to_bf16_kernel<<<cvtBlocks, 256, 0, stream>>>(x, abf, (int)ND);
  gemm_bf16_wmma_kernel<<<gemmBlocks, 256, 0, stream>>>(abf, wt_in, b_in, hbuf);

  // ------------------------------ GAT layers -------------------------------
  for (int i = 0; i < LAYERS; ++i) {
    f32_to_bf16_kernel<<<cvtBlocks, 256, 0, stream>>>(hbuf, abf, (int)ND);

    const float* identp = hbuf;
    if (i > 0) {
      gemm_bf16_wmma_kernel<<<gemmBlocks, 256, 0, stream>>>(
          abf, wtRes + (size_t)(i - 1) * DD2, bres + (size_t)(i - 1) * DIM, identbuf);
      identp = identbuf;
    }
    gemm_bf16_wmma_kernel<<<gemmBlocks, 256, 0, stream>>>(
        abf, wtL + (size_t)i * DD2, bl + (size_t)i * DIM, xlbuf);
    gemm_bf16_wmma_kernel<<<gemmBlocks, 256, 0, stream>>>(
        abf, wtR + (size_t)i * DD2, (const float*)nullptr, xrbuf);

    init_layer_kernel<<<cvtBlocks, 256, 0, stream>>>(aggbuf, lmaxb, denomb, colsumb,
                                                     (int)ND, (int)NH);

    edge_logits_kernel<<<edgeBlocks, 256, 0, stream>>>(
        srcp, dstp, xlbuf, xrbuf, att + (size_t)i * DIM, logitsb, lmaxb);
    edge_aggregate_kernel<<<edgeBlocks, 256, 0, stream>>>(
        srcp, dstp, xlbuf, logitsb, lmaxb, denomb, aggbuf);
    node_finalize_kernel<<<nodeBlocks, 256, 0, stream>>>(
        aggbuf, denomb, b_out + (size_t)i * DIM, colsumb);
    row_final_kernel<<<nodeBlocks, 256, 0, stream>>>(
        aggbuf, colsumb, identp, ln_g + (size_t)i * DIM, ln_b + (size_t)i * DIM,
        hbuf, (i < LAYERS - 1) ? 1 : 0);
  }
}
